// GNNPolicyNet_22737556865375
// MI455X (gfx1250) — matrix operand order repaired and checked by reading, not compiled, not run on hardware
//
#include <hip/hip_runtime.h>
#include <hip/hip_bf16.h>

#define NN   100000
#define EE   1600000
#define BBG  64
#define FINF 32
#define HH   128
#define GGD  20
#define AAC  6158
#define PPL  4
#define LLY  6
#define EPSF 1e-5f

typedef __bf16 bf16;
typedef bf16  v16bf __attribute__((ext_vector_type(16)));
typedef float v8f   __attribute__((ext_vector_type(8)));

// ---------------- gfx1250 async copy helpers ----------------
__device__ __forceinline__ unsigned lds_off_of(const void* p) {
  return (unsigned)(__SIZE_TYPE__)(__attribute__((address_space(3))) const char*)p;
}
__device__ __forceinline__ void async_g2l_b128(unsigned lds_byte_off, const void* gptr) {
  // GLOBAL_LOAD_ASYNC_TO_LDS_B128: VDST = per-lane LDS byte offset, VADDR = 64-bit addr
  asm volatile("global_load_async_to_lds_b128 %0, %1, off"
               :: "v"(lds_byte_off), "v"((unsigned long long)(__SIZE_TYPE__)gptr)
               : "memory");
}
__device__ __forceinline__ void wait_async0() {
  asm volatile("s_wait_asynccnt 0x0" ::: "memory");
}

// ---------------- WMMA fragment helpers ----------------
// 16-bit A fragment 16x32, contiguous k order used here (B matches): lane holds row M=lane%16;
// lanes 0-15 get K {0..15}, lanes 16-31 get K {16..31}.
__device__ __forceinline__ v16bf make_a_frag(const float* __restrict__ row,
                                             int koff, int lane, float scale) {
  const int base0 = (lane < 16) ? 0 : 16;
  const float* p0 = row + koff + base0;
  const float* p1 = row + koff + base0 + 8;
  v16bf a;
#pragma unroll
  for (int j = 0; j < 8; ++j) a[j] = (bf16)(p0[j] * scale);
#pragma unroll
  for (int j = 0; j < 8; ++j) a[8 + j] = (bf16)(p1[j] * scale);
  return a;
}

// ---------------- node encoder: h = relu(LN(x@W_ne + b)) ----------------
__global__ void __launch_bounds__(256) k_node_enc(
    const float* __restrict__ x, const float* __restrict__ W,
    const float* __restrict__ bias, const float* __restrict__ g,
    const float* __restrict__ be, float* __restrict__ h) {
  __shared__ __align__(32) bf16 Wt[HH * FINF];  // [n][k] col-major, 8KB
  const int tid = threadIdx.x;
  for (int i = tid; i < FINF * HH; i += 256) {
    int k = i >> 7, n = i & 127;          // W row-major (FIN,H)
    Wt[n * FINF + k] = (bf16)W[i];
  }
  __syncthreads();
  const int wave = tid >> 5, lane = tid & 31;
  const int tile = blockIdx.x * 8 + wave;
  if (tile >= NN / 16) return;
  const int rbase = tile * 16;
  const int arow  = rbase + (lane & 15);
  const int kstart = (lane < 16) ? 0 : 16;

  v16bf a = make_a_frag(x + (size_t)arow * FINF, 0, lane, 1.0f);
  v8f acc[8] = {};
#pragma unroll
  for (int t = 0; t < 8; ++t) {
    v16bf b = *(const v16bf*)(Wt + (size_t)(t * 16 + (lane & 15)) * FINF + kstart);
    acc[t] = __builtin_amdgcn_wmma_f32_16x16x32_bf16(false, a, false, b,
                                                     (short)0, acc[t], false, false);
  }

  float gv[8], bev[8], bsv[8];
#pragma unroll
  for (int t = 0; t < 8; ++t) {
    int col = t * 16 + (lane & 15);
    bsv[t] = bias[col]; gv[t] = g[col]; bev[t] = be[col];
  }
  float val[8][8];  // [r][t]
#pragma unroll
  for (int t = 0; t < 8; ++t)
#pragma unroll
    for (int r = 0; r < 8; ++r) val[r][t] = acc[t][r] + bsv[t];

  const int rowoff = (lane < 16) ? 0 : 8;
#pragma unroll
  for (int r = 0; r < 8; ++r) {
    float sfull = 0.f;
#pragma unroll
    for (int t = 0; t < 8; ++t) sfull += val[r][t];
#pragma unroll
    for (int m = 1; m < 16; m <<= 1) sfull += __shfl_xor(sfull, m, 32);
    float mean = sfull * (1.0f / HH);
    float vsum = 0.f;
#pragma unroll
    for (int t = 0; t < 8; ++t) { float d = val[r][t] - mean; vsum += d * d; }
#pragma unroll
    for (int m = 1; m < 16; m <<= 1) vsum += __shfl_xor(vsum, m, 32);
    float inv = rsqrtf(vsum * (1.0f / HH) + EPSF);
    int row = rbase + rowoff + r;
#pragma unroll
    for (int t = 0; t < 8; ++t) {
      int col = t * 16 + (lane & 15);
      float y = (val[r][t] - mean) * inv * gv[t] + bev[t];
      h[(size_t)row * HH + col] = fmaxf(y, 0.0f);
    }
  }
}

// ---------------- degree / scatter-add ----------------
__global__ void k_deg(const int* __restrict__ ei, float* __restrict__ deg) {
  int e = blockIdx.x * blockDim.x + threadIdx.x;
  if (e >= EE) return;
  atomicAdd(&deg[ei[EE + e]], 1.0f);
}

__global__ void k_scatter(const float* __restrict__ h, const int* __restrict__ ei,
                          float* __restrict__ agg) {
  int gid = blockIdx.x * blockDim.x + threadIdx.x;   // EE*32 threads
  int e = gid >> 5;
  if (e >= EE) return;
  int c4 = (gid & 31) << 2;
  int s = ei[e], d = ei[EE + e];
  float4 v = *(const float4*)(h + (size_t)s * HH + c4);
  float* ap = agg + (size_t)d * HH + c4;
  atomicAdd(ap + 0, v.x); atomicAdd(ap + 1, v.y);
  atomicAdd(ap + 2, v.z); atomicAdd(ap + 3, v.w);
}

// ---------------- fused SAGE layer (WMMA + LN + ReLU + residual) ----------------
__global__ void __launch_bounds__(256) k_sage(
    float* __restrict__ h, const float* __restrict__ agg,
    const float* __restrict__ deg, const float* __restrict__ Wl,
    const float* __restrict__ bl, const float* __restrict__ Wr,
    const float* __restrict__ lg, const float* __restrict__ lb) {
  __shared__ __align__(32) bf16  Wt[HH * 2 * HH];   // [n][k 0..255], 64KB bf16
  __shared__ __align__(16) float stage[32 * HH];    // 16KB async staging
  const int tid = threadIdx.x;
  const unsigned stage_base = lds_off_of(stage);

  // Stage Wl then Wr through LDS with gfx1250 async copies (ASYNCcnt path),
  // converting each 32-row chunk into the bf16 column-major B layout.
#pragma unroll 1
  for (int half = 0; half < 2; ++half) {
    const float* Wsrc = half ? Wr : Wl;
    const int kofs = half ? 128 : 0;
#pragma unroll 1
    for (int c = 0; c < 4; ++c) {
#pragma unroll
      for (int j = 0; j < 4; ++j) {
        unsigned off = (unsigned)(tid + j * 256) * 16u;   // 16B per lane per issue
        async_g2l_b128(stage_base + off,
                       (const char*)Wsrc + (size_t)c * 16384 + off);
      }
      wait_async0();
      __syncthreads();
      for (int i = tid; i < 32 * HH; i += 256) {
        int kl = i >> 7, n = i & 127;                     // stage row-major [kl][n]
        Wt[(size_t)n * 256 + kofs + c * 32 + kl] = (bf16)stage[i];
      }
      __syncthreads();
    }
  }

  const int wave = tid >> 5, lane = tid & 31;
  const int tile = blockIdx.x * 8 + wave;
  if (tile >= NN / 16) return;
  const int rbase = tile * 16;
  const int arow  = rbase + (lane & 15);
  const float rdeg = 1.0f / fmaxf(deg[arow], 1.0f);
  const int kstart = (lane < 16) ? 0 : 16;

  v8f acc[8] = {};
#pragma unroll
  for (int kb = 0; kb < 4; ++kb) {                 // (agg/deg) @ Wl
    v16bf a = make_a_frag(agg + (size_t)arow * HH, kb * 32, lane, rdeg);
#pragma unroll
    for (int t = 0; t < 8; ++t) {
      v16bf b = *(const v16bf*)(Wt + (size_t)(t * 16 + (lane & 15)) * 256 + kb * 32 + kstart);
      acc[t] = __builtin_amdgcn_wmma_f32_16x16x32_bf16(false, a, false, b,
                                                       (short)0, acc[t], false, false);
    }
  }
#pragma unroll
  for (int kb = 0; kb < 4; ++kb) {                 // h @ Wr
    v16bf a = make_a_frag(h + (size_t)arow * HH, kb * 32, lane, 1.0f);
#pragma unroll
    for (int t = 0; t < 8; ++t) {
      v16bf b = *(const v16bf*)(Wt + (size_t)(t * 16 + (lane & 15)) * 256 + 128 + kb * 32 + kstart);
      acc[t] = __builtin_amdgcn_wmma_f32_16x16x32_bf16(false, a, false, b,
                                                       (short)0, acc[t], false, false);
    }
  }

  float gv[8], bev[8], bsv[8];
#pragma unroll
  for (int t = 0; t < 8; ++t) {
    int col = t * 16 + (lane & 15);
    bsv[t] = bl[col]; gv[t] = lg[col]; bev[t] = lb[col];
  }
  float val[8][8];
#pragma unroll
  for (int t = 0; t < 8; ++t)
#pragma unroll
    for (int r = 0; r < 8; ++r) val[r][t] = acc[t][r] + bsv[t];

  const int rowoff = (lane < 16) ? 0 : 8;
#pragma unroll
  for (int r = 0; r < 8; ++r) {
    float sfull = 0.f;
#pragma unroll
    for (int t = 0; t < 8; ++t) sfull += val[r][t];
#pragma unroll
    for (int m = 1; m < 16; m <<= 1) sfull += __shfl_xor(sfull, m, 32);
    float mean = sfull * (1.0f / HH);
    float vsum = 0.f;
#pragma unroll
    for (int t = 0; t < 8; ++t) { float d = val[r][t] - mean; vsum += d * d; }
#pragma unroll
    for (int m = 1; m < 16; m <<= 1) vsum += __shfl_xor(vsum, m, 32);
    float inv = rsqrtf(vsum * (1.0f / HH) + EPSF);
    int row = rbase + rowoff + r;
#pragma unroll
    for (int t = 0; t < 8; ++t) {
      int col = t * 16 + (lane & 15);
      float y = fmaxf((val[r][t] - mean) * inv * gv[t] + bev[t], 0.0f);
      size_t idx = (size_t)row * HH + col;
      h[idx] = h[idx] + y;                          // residual
    }
  }
}

// ---------------- attention scores: one wave per node ----------------
__global__ void __launch_bounds__(256) k_attn(
    const float* __restrict__ h, const float* __restrict__ Wa1,
    const float* __restrict__ ba1, const float* __restrict__ Wa2,
    const float* __restrict__ ba2, float* __restrict__ s) {
  const int wave = threadIdx.x >> 5, lane = threadIdx.x & 31;
  const int n = blockIdx.x * 8 + wave;
  if (n >= NN) return;
  float a1 = 0.f, a2 = 0.f;
  const float* hr = h + (size_t)n * HH;
#pragma unroll 4
  for (int k = 0; k < HH; ++k) {
    float hv = hr[k];
    a1 = fmaf(hv, Wa1[k * 64 + lane], a1);
    a2 = fmaf(hv, Wa1[k * 64 + 32 + lane], a2);
  }
  float part = tanhf(a1 + ba1[lane]) * Wa2[lane]
             + tanhf(a2 + ba1[32 + lane]) * Wa2[32 + lane];
#pragma unroll
  for (int m = 1; m < 32; m <<= 1) part += __shfl_xor(part, m, 32);
  if (lane == 0) s[n] = part + ba2[0];
}

// ---------------- segment softmax + pooling ----------------
__device__ __forceinline__ unsigned enc_f32(float f) {
  unsigned u = __float_as_uint(f);
  return (u & 0x80000000u) ? ~u : (u | 0x80000000u);
}
__device__ __forceinline__ float dec_f32(unsigned u) {
  unsigned b = (u & 0x80000000u) ? (u & 0x7fffffffu) : ~u;
  return __uint_as_float(b);
}

__global__ void k_seg_max(const float* __restrict__ s, const int* __restrict__ batch,
                          unsigned* __restrict__ smax) {
  int n = blockIdx.x * blockDim.x + threadIdx.x;
  if (n >= NN) return;
  atomicMax(&smax[batch[n]], enc_f32(s[n]));
}

__global__ void k_seg_exp(const float* __restrict__ s, const int* __restrict__ batch,
                          const unsigned* __restrict__ smax, float* __restrict__ ssum,
                          float* __restrict__ expw) {
  int n = blockIdx.x * blockDim.x + threadIdx.x;
  if (n >= NN) return;
  int b = batch[n];
  float e = __expf(s[n] - dec_f32(smax[b]));
  expw[n] = e;
  atomicAdd(&ssum[b], e);
}

__global__ void k_pool(const float* __restrict__ h, const int* __restrict__ batch,
                       const float* __restrict__ expw, const float* __restrict__ ssum,
                       float* __restrict__ grep) {
  int gid = blockIdx.x * blockDim.x + threadIdx.x;   // NN*32 threads
  int n = gid >> 5;
  if (n >= NN) return;
  int c4 = (gid & 31) << 2;
  int b = batch[n];
  float w = expw[n] / ssum[b];
  float4 v = *(const float4*)(h + (size_t)n * HH + c4);
  float* gp = grep + (size_t)b * HH + c4;
  atomicAdd(gp + 0, v.x * w); atomicAdd(gp + 1, v.y * w);
  atomicAdd(gp + 2, v.z * w); atomicAdd(gp + 3, v.w * w);
}

// ---------------- gr = [graph_repr | relu(globals@Wg+bg)] ----------------
__global__ void k_gr(const float* __restrict__ grep, const float* __restrict__ glob,
                     const float* __restrict__ Wg, const float* __restrict__ bg,
                     float* __restrict__ gr) {
  int b = blockIdx.x, c = threadIdx.x;   // 128 threads
  gr[b * 256 + c] = grep[b * HH + c];
  float acc = bg[c];
#pragma unroll
  for (int g = 0; g < GGD; ++g) acc = fmaf(glob[b * GGD + g], Wg[g * HH + c], acc);
  gr[b * 256 + 128 + c] = fmaxf(acc, 0.0f);
}

// ---------------- generic small dense layer ----------------
__global__ void k_linear(const float* __restrict__ X, const float* __restrict__ W,
                         const float* __restrict__ bias, float* __restrict__ Y,
                         int M, int K, int Nc, int act) {
  int gid = blockIdx.x * blockDim.x + threadIdx.x;
  if (gid >= M * Nc) return;
  int m = gid / Nc, n = gid - m * Nc;
  float acc = bias[n];
  const float* xr = X + (size_t)m * K;
  for (int k = 0; k < K; ++k) acc = fmaf(xr[k], W[(size_t)k * Nc + n], acc);
  if (act == 1) acc = fmaxf(acc, 0.0f);
  else if (act == 2) acc = tanhf(acc);
  Y[(size_t)m * Nc + n] = acc;
}

// ---------------- launch ----------------
extern "C" void kernel_launch(void* const* d_in, const int* in_sizes, int n_in,
                              void* d_out, int out_size, void* d_ws, size_t ws_size,
                              hipStream_t stream) {
  (void)in_sizes; (void)n_in; (void)out_size; (void)ws_size;
  const float* x      = (const float*)d_in[0];
  const int*   ei     = (const int*)  d_in[1];
  const int*   batch  = (const int*)  d_in[2];
  const float* glob   = (const float*)d_in[3];
  const float* W_ne   = (const float*)d_in[4];
  const float* b_ne   = (const float*)d_in[5];
  const float* g_ne   = (const float*)d_in[6];
  const float* be_ne  = (const float*)d_in[7];
  const float* Wl     = (const float*)d_in[8];
  const float* bl     = (const float*)d_in[9];
  const float* Wr     = (const float*)d_in[10];
  const float* ln_g   = (const float*)d_in[11];
  const float* ln_b   = (const float*)d_in[12];
  const float* Wa1    = (const float*)d_in[13];
  const float* ba1    = (const float*)d_in[14];
  const float* Wa2    = (const float*)d_in[15];
  const float* ba2    = (const float*)d_in[16];
  const float* Wg     = (const float*)d_in[17];
  const float* bg     = (const float*)d_in[18];
  const float* Wpg    = (const float*)d_in[19];
  const float* bpg    = (const float*)d_in[20];
  const float* Wp1    = (const float*)d_in[21];
  const float* bp1    = (const float*)d_in[22];
  const float* Wp2    = (const float*)d_in[23];
  const float* bp2    = (const float*)d_in[24];
  const float* Wv1    = (const float*)d_in[25];
  const float* bv1    = (const float*)d_in[26];
  const float* Wv2    = (const float*)d_in[27];
  const float* bv2    = (const float*)d_in[28];
  const float* Wv3    = (const float*)d_in[29];
  const float* bv3    = (const float*)d_in[30];
  float* out = (float*)d_out;

  char* ws = (char*)d_ws;
  size_t off = 0;
  auto alloc = [&](size_t bytes) -> void* {
    void* p = ws + off;
    off = (off + bytes + 255) & ~(size_t)255;
    return p;
  };
  float*    h    = (float*)   alloc((size_t)NN * HH * 4);
  float*    agg  = (float*)   alloc((size_t)NN * HH * 4);
  float*    deg  = (float*)   alloc((size_t)NN * 4);
  float*    sbuf = (float*)   alloc((size_t)NN * 4);
  float*    expw = (float*)   alloc((size_t)NN * 4);
  unsigned* smax = (unsigned*)alloc((size_t)BBG * 4);
  float*    ssum = (float*)   alloc((size_t)BBG * 4);
  float*    grep = (float*)   alloc((size_t)BBG * HH * 4);
  float*    gr   = (float*)   alloc((size_t)BBG * 2 * HH * 4);
  float*    pbuf = (float*)   alloc((size_t)BBG * HH * 4);
  float*    qbuf = (float*)   alloc((size_t)BBG * HH * 4);
  float*    v1   = (float*)   alloc((size_t)BBG * HH * 4);
  float*    v2   = (float*)   alloc((size_t)BBG * 64 * 4);

  hipMemsetAsync(deg,  0, (size_t)NN * 4, stream);
  hipMemsetAsync(smax, 0, (size_t)BBG * 4, stream);      // enc(-inf) <= 0x007FFFFF
  hipMemsetAsync(ssum, 0, (size_t)BBG * 4, stream);
  hipMemsetAsync(grep, 0, (size_t)BBG * HH * 4, stream);

  const int tiles = NN / 16;                              // 6250, exact
  const int gemm_blocks = (tiles + 7) / 8;                // 8 waves / block

  k_node_enc<<<gemm_blocks, 256, 0, stream>>>(x, W_ne, b_ne, g_ne, be_ne, h);
  k_deg<<<(EE + 255) / 256, 256, 0, stream>>>(ei, deg);

  for (int i = 0; i < LLY; ++i) {
    hipMemsetAsync(agg, 0, (size_t)NN * HH * 4, stream);
    k_scatter<<<(EE * 32) / 256, 256, 0, stream>>>(h, ei, agg);
    k_sage<<<gemm_blocks, 256, 0, stream>>>(h, agg, deg,
        Wl + (size_t)i * HH * HH, bl + (size_t)i * HH,
        Wr + (size_t)i * HH * HH, ln_g + (size_t)i * HH, ln_b + (size_t)i * HH);
  }

  k_attn<<<(NN + 7) / 8, 256, 0, stream>>>(h, Wa1, ba1, Wa2, ba2, sbuf);
  k_seg_max<<<(NN + 255) / 256, 256, 0, stream>>>(sbuf, batch, smax);
  k_seg_exp<<<(NN + 255) / 256, 256, 0, stream>>>(sbuf, batch, smax, ssum, expw);
  k_pool<<<(NN * 32) / 256, 256, 0, stream>>>(h, batch, expw, ssum, grep);
  k_gr<<<BBG, 128, 0, stream>>>(grep, glob, Wg, bg, gr);

  // policy head
  k_linear<<<(BBG * HH + 255) / 256, 256, 0, stream>>>(gr,   Wpg, bpg, pbuf, BBG, 2 * HH, HH, 0);
  k_linear<<<(BBG * HH + 255) / 256, 256, 0, stream>>>(pbuf, Wp1, bp1, qbuf, BBG, HH, HH, 1);
  k_linear<<<(BBG * AAC + 255) / 256, 256, 0, stream>>>(qbuf, Wp2, bp2, out,  BBG, HH, AAC, 0);
  // value head
  k_linear<<<(BBG * HH + 255) / 256, 256, 0, stream>>>(gr, Wv1, bv1, v1, BBG, 2 * HH, HH, 1);
  k_linear<<<(BBG * 64 + 255) / 256, 256, 0, stream>>>(v1, Wv2, bv2, v2, BBG, HH, 64, 1);
  k_linear<<<1, 256, 0, stream>>>(v2, Wv3, bv3, out + (size_t)BBG * AAC, BBG, 64, PPL, 2);
}
